// Trittention_27273042329830
// MI455X (gfx1250) — matrix operand order repaired and checked by reading, not compiled, not run on hardware
//
#include <hip/hip_runtime.h>
#include <hip/hip_bf16.h>

// Trittention forward for MI455X (gfx1250, wave32, WMMA + TDM).
// Dominant cost: per-(b,n,q) 128x128x64 GEMM  S = (k1 * diag(qv)) @ k2^T,
// via v_wmma_f32_16x16x32_bf16 (f32 accumulate). Masked softmax marginals
// (row/col exp-sums) are accumulated straight from WMMA fragments; wave w
// only visits tiles tt in [w, ceil(q/16)) since valid region is s < t < q.
// k2 (bf16) is staged into LDS by the Tensor Data Mover (tensor_load_to_lds
// + s_wait_tensorcnt). All reductions are order-deterministic.

#define BS 2
#define TS 128
#define DM 512
#define NH 8
#define DH 64

typedef __attribute__((ext_vector_type(16))) __bf16 v16bf;
typedef __attribute__((ext_vector_type(8)))  float  v8f;
typedef __attribute__((ext_vector_type(4)))  unsigned int v4u;
typedef __attribute__((ext_vector_type(4)))  int    v4i;
typedef __attribute__((ext_vector_type(8)))  int    v8i;

union FragU {
    uint4 u2[2];
    v16bf v;
};

// ---------------------------------------------------------------------------
// TDM: DMA a contiguous 8192-element (16 KB) bf16 tile Global -> LDS.
// D# layout per CDNA5 ISA ch.8 (group0: count/lds_addr/global_addr/type=2;
// group1: data_size=2B, tensor_dim0=tile_dim0=8192, tensor_dim1=tile_dim1=1).
// ---------------------------------------------------------------------------
__device__ __forceinline__ void tdm_load_bf16_16k(unsigned int lds_off,
                                                  unsigned long long gaddr) {
    v4u g0;
    g0[0] = 1u;                                        // count=1 (valid user D#)
    g0[1] = lds_off;                                   // lds_addr (bytes)
    g0[2] = (unsigned int)(gaddr & 0xFFFFFFFFu);       // global_addr[31:0]
    g0[3] = (unsigned int)((gaddr >> 32) & 0x01FFFFFFu)
          | (2u << 30);                                // global_addr[56:32], type=2
    v8i g1;
    g1[0] = (int)(1u << 16);                           // data_size=1 (2 bytes)
    g1[1] = (int)(8192u << 16);                        // tensor_dim0 lo16 -> [31:16]
    g1[2] = (int)(1u << 16);                           // tensor_dim0 hi=0; tensor_dim1=1
    g1[3] = (int)(8192u << 16);                        // tensor_dim1 hi=0; tile_dim0=8192
    g1[4] = 1;                                         // tile_dim1=1, tile_dim2=0
    g1[5] = 8192;                                      // tensor_dim0_stride lo32
    g1[6] = (int)(8192u << 16);                        // stride0 hi=0; stride1 lo16
    g1[7] = 0;                                         // stride1 hi
    v4i gz = {0, 0, 0, 0};
#if __clang_major__ >= 23
    v8i gz8 = {0, 0, 0, 0, 0, 0, 0, 0};
    __builtin_amdgcn_tensor_load_to_lds(g0, g1, gz, gz, gz8, 0);
#else
    __builtin_amdgcn_tensor_load_to_lds(g0, g1, gz, gz, 0);
#endif
}

// ---------------------------------------------------------------------------
// Kernel 1: projections k1,k2,q,va,vb  (f32 VALU, ~0.67 GFLOP, L2-resident)
// outputs laid out [b][n][p][h]; sel==1 additionally emits bf16 k2 for TDM.
// ---------------------------------------------------------------------------
__global__ __launch_bounds__(256) void proj_kernel(
    const float* __restrict__ x,
    const float* __restrict__ WK1, const float* __restrict__ WK2,
    const float* __restrict__ WQ,  const float* __restrict__ WV12,
    const float* __restrict__ bK1, const float* __restrict__ bK2,
    const float* __restrict__ bQ,
    float* __restrict__ k1, float* __restrict__ k2, float* __restrict__ qm,
    float* __restrict__ va, float* __restrict__ vb,
    __bf16* __restrict__ k2bf)
{
    const int tid   = threadIdx.x;
    const int sel   = blockIdx.x / (BS * NH);
    const int bn    = blockIdx.x % (BS * NH);
    const int b     = bn >> 3;
    const int n     = bn & 7;
    const int chunk = blockIdx.y;               // 0..3 -> 32 p rows each

    const float* W;
    const float* bias;
    float* outp;
    switch (sel) {
        case 0:  W = WK1  + (size_t)n * DM * DH;               bias = bK1 + n * DH; outp = k1; break;
        case 1:  W = WK2  + (size_t)n * DM * DH;               bias = bK2 + n * DH; outp = k2; break;
        case 2:  W = WQ   + (size_t)n * DM * DH;               bias = bQ  + n * DH; outp = qm; break;
        case 3:  W = WV12 + (size_t)n * 2 * DM * DH;           bias = nullptr;      outp = va; break;
        default: W = WV12 + (size_t)n * 2 * DM * DH + (size_t)DM * DH;
                 bias = nullptr;                                                    outp = vb; break;
    }

    __shared__ float xs[8][DM + 8];   // padded to dodge bank conflicts
    const int h  = tid & 63;
    const int pp = tid >> 6;          // 0..3, each handles 2 local p rows

    for (int sub = 0; sub < 4; ++sub) {
        const int pstart = chunk * 32 + sub * 8;
        for (int i = tid; i < 8 * DM; i += 256) {
            const int lp = i >> 9;
            const int d  = i & (DM - 1);
            xs[lp][d] = x[((size_t)(b * TS) + pstart + lp) * DM + d];
        }
        __syncthreads();

        const int lp0 = pp * 2, lp1 = pp * 2 + 1;
        float acc0 = 0.f, acc1 = 0.f;
        for (int d = 0; d < DM; ++d) {
            const float w = W[(size_t)d * DH + h];
            acc0 += xs[lp0][d] * w;
            acc1 += xs[lp1][d] * w;
        }
        const float bb = bias ? bias[h] : 0.f;
        const size_t ob = ((size_t)bn * TS + pstart) * DH + h;
        const float r0 = acc0 + bb, r1 = acc1 + bb;
        outp[ob + (size_t)lp0 * DH] = r0;
        outp[ob + (size_t)lp1 * DH] = r1;
        if (sel == 1) {
            k2bf[ob + (size_t)lp0 * DH] = (__bf16)r0;
            k2bf[ob + (size_t)lp1 * DH] = (__bf16)r1;
        }
        __syncthreads();
    }
}

// ---------------------------------------------------------------------------
// Kernel 2: attention core. One block per (b,n,q); 8 waves, wave w owns
// s-rows [16w,16w+16) and visits only t-tiles [w, ceil(q/16)).
// ---------------------------------------------------------------------------
__global__ __launch_bounds__(256) void attn_kernel(
    const float* __restrict__ k1, const __bf16* __restrict__ k2bf,
    const float* __restrict__ qm, const float* __restrict__ va,
    const float* __restrict__ vb, const float* __restrict__ bV12,
    float* __restrict__ zout)
{
    const int tid = threadIdx.x;
    const int bnq = blockIdx.x;
    const int qi  = bnq & (TS - 1);
    const int bn  = bnq >> 7;
    const int n   = bn & (NH - 1);
    const int b   = bn >> 3;
    const size_t bnoff = (size_t)bn * TS * DH;

    // q<2: everything masked, all softmax mass on the sink (vf==0) -> z = 0
    if (qi < 2) {
        if (tid < DH) zout[((size_t)(b * TS + qi) * NH + n) * DH + tid] = 0.0f;
        return;
    }

    __shared__ float  qv[DH];
    __shared__ __bf16 Ak[TS][DH];        // k1 * qv  (bf16)
    __shared__ __bf16 Bk[TS][DH];        // k2       (bf16), filled by TDM
    __shared__ float  rowsum[TS];
    __shared__ float  colsum[TS];
    __shared__ float  colpart[16][TS];   // row = 2*wave + laneHi
    __shared__ float  part[4][DH];

    if (tid == 0) __builtin_prefetch(k1 + bnoff, 0, 1);
    if (tid < DH) qv[tid] = qm[bnoff + (size_t)qi * DH + tid];
    for (int i = tid; i < 16 * TS; i += 256) ((float*)colpart)[i] = 0.f;

    // Wave 0 kicks off the TDM for the k2 tile (16 KB, Global -> LDS Bk).
    if (tid < 32) {
        tdm_load_bf16_16k((unsigned int)(uintptr_t)(void*)&Bk[0][0],
                          (unsigned long long)(uintptr_t)(const void*)(k2bf + bnoff));
    }
    __syncthreads();                      // qv visible for Ak staging

    for (int i = tid * 32, e = i + 32; i < e; ++i) {
        const int h = i & (DH - 1);
        ((__bf16*)Ak)[i] = (__bf16)(k1[bnoff + i] * qv[h]);
    }
    if (tid < 32) __builtin_amdgcn_s_wait_tensorcnt(0);
    __syncthreads();                      // Ak (ds) + Bk (TDM) both ready

    const int  wave = tid >> 5;
    const int  lane = tid & 31;
    const int  lm   = lane & 15;
    const bool hiL  = lane >= 16;
    const int  s0   = wave * 16;

    // A fragments (16x32 bf16 layout): lanes 0-15 row M=lane K{0..7,16..23},
    // lanes 16-31 row M=lane-16 K{8..15,24..31}; +32 for second K-step.
    FragU a0, a1;
    {
        const __bf16* rowp = &Ak[s0 + lm][0];
        const int c0 = hiL ? 8 : 0;
        a0.u2[0] = *(const uint4*)(rowp + c0);
        a0.u2[1] = *(const uint4*)(rowp + c0 + 16);
        a1.u2[0] = *(const uint4*)(rowp + c0 + 32);
        a1.u2[1] = *(const uint4*)(rowp + c0 + 48);
    }

    float rs_acc[8];
#pragma unroll
    for (int r = 0; r < 8; ++r) rs_acc[r] = 0.f;

    const int ttq = (qi + 15) >> 4;       // tiles with any tg < qi
    for (int tt = wave; tt < ttq; ++tt) { // tt < wave has no sg < tg pairs
        // B fragment (32x16 bf16): lane n=lm holds K{0..7,16..23} (lanes<16) or
        // K{8..15,24..31} (lanes>=16); B[k][n] = Bk[t0+n][k].
        FragU b0, b1;
        const __bf16* rowp = &Bk[tt * 16 + lm][0];
        const int c0 = hiL ? 8 : 0;
        b0.u2[0] = *(const uint4*)(rowp + c0);
        b0.u2[1] = *(const uint4*)(rowp + c0 + 16);
        b1.u2[0] = *(const uint4*)(rowp + c0 + 32);
        b1.u2[1] = *(const uint4*)(rowp + c0 + 48);

        v8f c = {0.f, 0.f, 0.f, 0.f, 0.f, 0.f, 0.f, 0.f};
        c = __builtin_amdgcn_wmma_f32_16x16x32_bf16(false, a0.v, false, b0.v,
                                                    (short)0, c, false, false);
        c = __builtin_amdgcn_wmma_f32_16x16x32_bf16(false, a1.v, false, b1.v,
                                                    (short)0, c, false, false);

        // C/D layout: VGPR r -> lanes0-15: M=r, lanes16-31: M=8+r; N = lane&15
        const int tg = tt * 16 + lm;
        float colacc = 0.f;
#pragma unroll
        for (int r = 0; r < 8; ++r) {
            const int sg = s0 + r + (hiL ? 8 : 0);
            const bool valid = (sg < tg) && (tg < qi);      // s < t < q
            const float p = valid ? __expf(c[r] * 0.015625f) : 0.0f;
            colacc += p;
            rs_acc[r] += p;               // per-lane partial; reduced after loop
        }
        colpart[2 * wave + (hiL ? 1 : 0)][tg] = colacc;     // unique writer
    }

    // One 16-lane butterfly per row AFTER the tile loop (not per tile).
#pragma unroll
    for (int r = 0; r < 8; ++r) {
        float pr = rs_acc[r];
        pr += __shfl_xor(pr, 1, 32);
        pr += __shfl_xor(pr, 2, 32);
        pr += __shfl_xor(pr, 4, 32);
        pr += __shfl_xor(pr, 8, 32);
        if (lm == 0) rowsum[s0 + r + (hiL ? 8 : 0)] = pr;
    }
    __syncthreads();

    if (tid < TS) {
        float s = 0.f;
        for (int w = 0; w < 16; ++w) s += colpart[w][tid];
        colsum[tid] = s;
    }
    __syncthreads();

    float Z = 0.f;                       // identical serial order on all threads
    for (int s = 0; s < TS; ++s) Z += rowsum[s];
    const float invZ = 1.0f / Z;         // sink weight exp(-1000) == 0 in fp32

    // z[h] = (sum_s rowsum[s]*va[s,h] + sum_t colsum[t]*vb[t,h]) / Z + b_V12[h]
    const int c4 = tid >> 6;
    const int h  = tid & 63;
    const float* vap = va + bnoff + h;
    const float* vbp = vb + bnoff + h;
    float acc = 0.f;
    for (int s = c4 * 32, e = s + 32; s < e; ++s)
        acc += rowsum[s] * vap[(size_t)s * DH] + colsum[s] * vbp[(size_t)s * DH];
    part[c4][h] = acc;
    __syncthreads();

    if (tid < DH) {
        const float tot = part[0][tid] + part[1][tid] + part[2][tid] + part[3][tid];
        zout[((size_t)(b * TS + qi) * NH + n) * DH + tid] =
            tot * invZ + bV12[n * DH + tid];
    }
}

// ---------------------------------------------------------------------------
// Kernel 3: output projection  out[b,p,:] = z[b,p,:] @ W_O(512,512) + b_O
// ---------------------------------------------------------------------------
__global__ __launch_bounds__(256) void oproj_kernel(
    const float* __restrict__ z, const float* __restrict__ WO,
    const float* __restrict__ bO, float* __restrict__ out)
{
    const int tid = threadIdx.x;
    const int row = blockIdx.x;                 // b*TS + p
    __shared__ float zs[DM];
    zs[tid]       = z[(size_t)row * DM + tid];
    zs[tid + 256] = z[(size_t)row * DM + tid + 256];
    __syncthreads();

    float a0 = 0.f, a1 = 0.f;
    for (int k = 0; k < DM; ++k) {
        const float zk = zs[k];
        a0 += zk * WO[(size_t)k * DM + tid];
        a1 += zk * WO[(size_t)k * DM + tid + 256];
    }
    out[(size_t)row * DM + tid]       = a0 + bO[tid];
    out[(size_t)row * DM + tid + 256] = a1 + bO[tid + 256];
}

// ---------------------------------------------------------------------------
extern "C" void kernel_launch(void* const* d_in, const int* in_sizes, int n_in,
                              void* d_out, int out_size, void* d_ws, size_t ws_size,
                              hipStream_t stream) {
    const float* x    = (const float*)d_in[0];
    const float* WK1  = (const float*)d_in[1];
    const float* WK2  = (const float*)d_in[2];
    const float* WQ   = (const float*)d_in[3];
    const float* WV12 = (const float*)d_in[4];
    const float* WO   = (const float*)d_in[5];
    const float* bK1  = (const float*)d_in[6];
    const float* bK2  = (const float*)d_in[7];
    const float* bQ   = (const float*)d_in[8];
    const float* bV12 = (const float*)d_in[9];
    const float* bO   = (const float*)d_in[10];

    const size_t HEADSZ = (size_t)BS * NH * TS * DH;  // 131072 floats
    float* ws = (float*)d_ws;
    float*  k1   = ws;
    float*  k2   = ws + 1 * HEADSZ;
    float*  qm   = ws + 2 * HEADSZ;
    float*  va   = ws + 3 * HEADSZ;
    float*  vb   = ws + 4 * HEADSZ;
    float*  zb   = ws + 5 * HEADSZ;
    __bf16* k2bf = (__bf16*)(ws + 6 * HEADSZ);        // 256 KB bf16 copy of k2

    dim3 gp(5 * BS * NH, 4);
    proj_kernel<<<gp, 256, 0, stream>>>(x, WK1, WK2, WQ, WV12, bK1, bK2, bQ,
                                        k1, k2, qm, va, vb, k2bf);
    attn_kernel<<<BS * NH * TS, 256, 0, stream>>>(k1, k2bf, qm, va, vb, bV12, zb);
    oproj_kernel<<<BS * TS, 256, 0, stream>>>(zb, WO, bO, (float*)d_out);
}
